// SynthesizerAttention_30107720745074
// MI455X (gfx1250) — matrix operand order repaired and checked by reading, not compiled, not run on hardware
//
#include <hip/hip_runtime.h>

// ---------------------------------------------------------------------------
// Synthesizer attention for MI455X (gfx1250), wave32 + WMMA bf16.
//   1. convert x, w1_w, value_w, proj_w  f32 -> bf16
//   2. transpose w2 [H,hd,M] -> w2t [H,M,hd] bf16
//   3. GEMM k  = relu(x@w1^T + b1) -> bf16 [B*T, C]     (async-LDS staged B)
//   4. GEMM v  = x@value^T + bv    -> bf16 vT [B,H,hd,T]
//   5. fused flash-style attention (512MB score tensor never materialized)
//   6. GEMM out = attnout@proj^T + bp -> f32 d_out
// ---------------------------------------------------------------------------

typedef __attribute__((ext_vector_type(16))) __bf16 v16bf;
typedef __attribute__((ext_vector_type(8)))  float  v8f;

union FragU {
  uint4 q[2];   // two 16-byte halves
  v16bf bf;     // 16 bf16 = 8 VGPRs
};

__device__ __forceinline__ unsigned short f2bf_bits(float f) {
  unsigned int u = __builtin_bit_cast(unsigned int, f);
  u += 0x7FFFu + ((u >> 16) & 1u);           // round-to-nearest-even
  return (unsigned short)(u >> 16);
}

// two 8-bf16 runs at p and p+16 (two b128 loads) -> B-fragment
__device__ __forceinline__ void load_frag16(FragU& f, const unsigned short* p) {
  f.q[0] = *(const uint4*)(p);
  f.q[1] = *(const uint4*)(p + 16);
}

// gfx1250 async copy: 16B global -> LDS, tracked by ASYNCcnt (ISA ch.10/15.18)
__device__ __forceinline__ void async_copy_b128(unsigned lds_off, const void* g) {
  unsigned long long ga = (unsigned long long)(size_t)g;
  asm volatile("global_load_async_to_lds_b128 %0, %1, off"
               :: "v"(lds_off), "v"(ga) : "memory");
}
__device__ __forceinline__ void wait_asynccnt0() {
  asm volatile("s_wait_asynccnt 0x0" ::: "memory");
}

#define BDIM 2
#define TDIM 2048
#define CDIM 1024
#define HDIM 16
#define HD   64
#define BT   (BDIM * TDIM)

// ---------------------------------------------------------------- prep ----
__global__ __launch_bounds__(256) void cvt_f32_bf16(const float* __restrict__ in,
                                                    unsigned short* __restrict__ out,
                                                    int n) {
  int i = blockIdx.x * blockDim.x + threadIdx.x;
  if (i < n) out[i] = f2bf_bits(in[i]);
}

// w2 [H][hd][M] f32  ->  w2t [H][M][hd] bf16
__global__ __launch_bounds__(256) void transpose_w2(const float* __restrict__ w2,
                                                    unsigned short* __restrict__ w2t) {
  int i = blockIdx.x * blockDim.x + threadIdx.x;
  if (i >= HDIM * TDIM * HD) return;
  int h   = i / (TDIM * HD);
  int rem = i % (TDIM * HD);
  int m   = rem / HD;
  int d   = rem % HD;
  w2t[i] = f2bf_bits(w2[((size_t)h * HD + d) * TDIM + m]);
}

// ---------------------------------------------------------------- GEMM ----
// C[m,n] = act( sum_k A[m,k]*W[n,k] + bias[n] )
// 256 thr = 8 waves; block tile 128(M) x 64(N); k-step 64.
// B (weight) tile 64n x 64k bf16 staged in LDS by two block-wide
// GLOBAL_LOAD_ASYNC_TO_LDS_B128 issues per k-step, double-buffered on
// ASYNCcnt (one barrier per 64-k step, 8 WMMAs/wave per barrier).
// B-fragments are batch-loaded (4 at a time) before the WMMA burst so the
// compiler emits one dscnt wait per 4 WMMAs.  A-fragments register-prefetched.
// mode 0: relu -> bf16 row-major.  mode 1: bf16 -> vT[b][h][d][t].
// mode 2: f32 row-major (final proj).
__global__ __launch_bounds__(256) void gemm_xWt(const unsigned short* __restrict__ A,
                                                const unsigned short* __restrict__ W,
                                                const float* __restrict__ bias,
                                                void* __restrict__ out, int mode) {
  __shared__ __align__(16) unsigned short bsm[2][64 * 64];  // [n][k64], 8KB each

  const int tid  = threadIdx.x;
  const int lane = tid & 31;
  const int wv   = tid >> 5;
  const int half = lane >> 4;
  const int ln   = lane & 15;
  const int m0   = blockIdx.x * 128 + wv * 16;
  const int n0   = blockIdx.y * 64;

  // staging map (two issues per 64-k step):
  //   issue 0: rows  0..31, issue 1: rows 32..63;  n = tid>>3, chunk = tid&7
  const int sn = tid >> 3, sc = tid & 7;
  const unsigned short* bsrc0 = W + (size_t)(n0 + sn) * CDIM + sc * 8;
  const unsigned short* bsrc1 = W + (size_t)(n0 + 32 + sn) * CDIM + sc * 8;
  const unsigned lds00 = (unsigned)(size_t)&bsm[0][sn * 64 + sc * 8];
  const unsigned lds01 = lds00 + 32 * 64 * 2;        // rows 32..63 of buf 0
  const unsigned lds10 = (unsigned)(size_t)&bsm[1][sn * 64 + sc * 8];
  const unsigned lds11 = lds10 + 32 * 64 * 2;

  v8f acc[4];
  for (int i = 0; i < 4; ++i)
    for (int j = 0; j < 8; ++j) acc[i][j] = 0.f;

  const unsigned short* arow = A + (size_t)(m0 + ln) * CDIM;

  // prologue: stage first B tile, load first two A fragments
  async_copy_b128(lds00, bsrc0);
  async_copy_b128(lds01, bsrc1);
  FragU af0, af1;
  af0.q[0] = *(const uint4*)(arow + half * 8);
  af0.q[1] = *(const uint4*)(arow + 16 + half * 8);
  af1.q[0] = *(const uint4*)(arow + 32 + half * 8);
  af1.q[1] = *(const uint4*)(arow + 48 + half * 8);

  for (int k0 = 0; k0 < CDIM; k0 += 64) {
    const int cur = (k0 >> 6) & 1;
    wait_asynccnt0();      // my wave's async writes for this tile done
    __syncthreads();       // all waves' writes done; other buffer free
    if (k0 + 64 < CDIM) {
      async_copy_b128(cur ? lds00 : lds10, bsrc0 + k0 + 64);
      async_copy_b128(cur ? lds01 : lds11, bsrc1 + k0 + 64);
    }

    FragU a0c = af0, a1c = af1;   // consume current A frags, prefetch next
    if (k0 + 64 < CDIM) {
      af0.q[0] = *(const uint4*)(arow + k0 + 64 + half * 8);
      af0.q[1] = *(const uint4*)(arow + k0 + 80 + half * 8);
      af1.q[0] = *(const uint4*)(arow + k0 + 96 + half * 8);
      af1.q[1] = *(const uint4*)(arow + k0 + 112 + half * 8);
    }

    const unsigned short* bb = &bsm[cur][0];
#pragma unroll
    for (int kk = 0; kk < 2; ++kk) {
      // batch all 4 B-fragments, then the WMMA burst
      FragU bfr[4];
#pragma unroll
      for (int ns = 0; ns < 4; ++ns)
        load_frag16(bfr[ns], bb + (ns * 16 + ln) * 64 + kk * 32 + half * 8);
      const FragU& aa = kk ? a1c : a0c;
#pragma unroll
      for (int ns = 0; ns < 4; ++ns)
        acc[ns] = __builtin_amdgcn_wmma_f32_16x16x32_bf16(
            false, aa.bf, false, bfr[ns].bf, (short)0, acc[ns], false, false);
    }
  }

#pragma unroll
  for (int ns = 0; ns < 4; ++ns) {
    const int gn = n0 + ns * 16 + ln;
    const float bv = bias[gn];
#pragma unroll
    for (int r = 0; r < 8; ++r) {
      float v = acc[ns][r] + bv;
      const int gm = m0 + r + half * 8;
      if (mode == 0) {
        v = fmaxf(v, 0.f);
        ((unsigned short*)out)[(size_t)gm * CDIM + gn] = f2bf_bits(v);
      } else if (mode == 1) {
        const int b = gm >> 11, t = gm & (TDIM - 1);
        const int h = gn >> 6,  d = gn & (HD - 1);
        ((unsigned short*)out)[(((size_t)b * HDIM + h) * HD + d) * TDIM + t] =
            f2bf_bits(v);
      } else {
        ((float*)out)[(size_t)gm * CDIM + gn] = v;
      }
    }
  }
}

// ----------------------------------------------------- fused attention ----
// Grid (T/64, H, B), 128 thr = 4 waves; wave handles 16 query rows.
// Software-pipelined: V-fragments issued before score WMMAs (complete
// during softmax), next key tile's w2t fragments prefetched while the
// current tile's softmax runs.
__global__ __launch_bounds__(128) void synth_attn(
    const unsigned short* __restrict__ Kmat,  // [BT, C] bf16
    const unsigned short* __restrict__ W2T,   // [H, M, hd] bf16
    const float* __restrict__ b2,             // [M]
    const unsigned short* __restrict__ VT,    // [B, H, hd, T] bf16
    unsigned short* __restrict__ O) {         // [BT, C] bf16
  __shared__ __align__(16) unsigned short pls[4][16 * 32];

  const int lane = threadIdx.x & 31;
  const int wv   = threadIdx.x >> 5;
  const int half = lane >> 4;
  const int ln   = lane & 15;
  const int b    = blockIdx.z;
  const int h    = blockIdx.y;
  const int q0   = blockIdx.x * 64 + wv * 16;

  // A-fragments of k for this wave's 16 queries: K-dim = hd = 64 -> 2 frags
  const unsigned short* krow =
      Kmat + (size_t)(b * TDIM + q0 + ln) * CDIM + h * HD;
  FragU a0, a1;
  a0.q[0] = *(const uint4*)(krow + half * 8);
  a0.q[1] = *(const uint4*)(krow + 16 + half * 8);
  a1.q[0] = *(const uint4*)(krow + 32 + half * 8);
  a1.q[1] = *(const uint4*)(krow + 48 + half * 8);

  v8f oacc[4];
  for (int i = 0; i < 4; ++i)
    for (int j = 0; j < 8; ++j) oacc[i][j] = 0.f;
  float run_max[8], run_sum[8];
#pragma unroll
  for (int r = 0; r < 8; ++r) { run_max[r] = -1e30f; run_sum[r] = 0.f; }

  const unsigned short* w2h   = W2T + (size_t)h * TDIM * HD;
  const unsigned short* vbase = VT + (size_t)(b * HDIM + h) * HD * TDIM;
  unsigned short* pl = &pls[wv][0];

  const int qmax = q0 + 15;

  // preload ns=0 w2t fragments for first key tile
  FragU nb0, nb1;
  {
    const unsigned short* brow = w2h + (size_t)ln * HD + half * 8;
    load_frag16(nb0, brow);
    load_frag16(nb1, brow + 32);
  }

  for (int j0 = 0; j0 <= qmax; j0 += 32) {
    // current tile fragments: ns=0 from prefetch, ns=1 loaded now
    FragU c00 = nb0, c01 = nb1, c10, c11;
    {
      const unsigned short* brow =
          w2h + (size_t)(j0 + 16 + ln) * HD + half * 8;
      load_frag16(c10, brow);
      load_frag16(c11, brow + 32);
    }
    // issue V-fragment loads early; they complete during softmax
    FragU bv[4];
#pragma unroll
    for (int ds = 0; ds < 4; ++ds)
      load_frag16(bv[ds], vbase + (size_t)(ds * 16 + ln) * TDIM + j0 + half * 8);

    v8f s[2];
    for (int i = 0; i < 2; ++i)
      for (int j = 0; j < 8; ++j) s[i][j] = 0.f;
    s[0] = __builtin_amdgcn_wmma_f32_16x16x32_bf16(
        false, a0.bf, false, c00.bf, (short)0, s[0], false, false);
    s[0] = __builtin_amdgcn_wmma_f32_16x16x32_bf16(
        false, a1.bf, false, c01.bf, (short)0, s[0], false, false);
    s[1] = __builtin_amdgcn_wmma_f32_16x16x32_bf16(
        false, a0.bf, false, c10.bf, (short)0, s[1], false, false);
    s[1] = __builtin_amdgcn_wmma_f32_16x16x32_bf16(
        false, a1.bf, false, c11.bf, (short)0, s[1], false, false);

    // prefetch next tile's ns=0 fragments (guarded: stays in-bounds)
    if (j0 + 32 <= qmax) {
      const unsigned short* brow =
          w2h + (size_t)(j0 + 32 + ln) * HD + half * 8;
      load_frag16(nb0, brow);
      load_frag16(nb1, brow + 32);
    }

    // bias + causal mask
#pragma unroll
    for (int ns = 0; ns < 2; ++ns) {
      const int col = j0 + ns * 16 + ln;
      const float bb = b2[col];
#pragma unroll
      for (int r = 0; r < 8; ++r) {
        const int row = q0 + r + half * 8;
        const float v = s[ns][r] + bb;
        s[ns][r] = (col <= row) ? v : -1e30f;
      }
    }

    // online softmax per row (rows live across the 16 lanes of one half)
#pragma unroll
    for (int r = 0; r < 8; ++r) {
      float mx = fmaxf(s[0][r], s[1][r]);
      for (int msk = 8; msk >= 1; msk >>= 1) mx = fmaxf(mx, __shfl_xor(mx, msk));
      const float newm  = fmaxf(run_max[r], mx);
      const float scale = __expf(run_max[r] - newm);
      run_max[r] = newm;
      const float p0 = __expf(s[0][r] - newm);
      const float p1 = __expf(s[1][r] - newm);
      s[0][r] = p0; s[1][r] = p1;
      float sm = p0 + p1;
      for (int msk = 8; msk >= 1; msk >>= 1) sm += __shfl_xor(sm, msk);
      run_sum[r] = run_sum[r] * scale + sm;
      oacc[0][r] *= scale; oacc[1][r] *= scale;
      oacc[2][r] *= scale; oacc[3][r] *= scale;
    }

    // P: f32 D-layout -> bf16 A-layout via per-wave LDS tile (DS in-order)
#pragma unroll
    for (int r = 0; r < 8; ++r) {
      const int row = r + half * 8;
      pl[row * 32 + ln]      = f2bf_bits(s[0][r]);
      pl[row * 32 + 16 + ln] = f2bf_bits(s[1][r]);
    }
    FragU ap;
    const unsigned short* prow = pl + (size_t)ln * 32;
    ap.q[0] = *(const uint4*)(prow + half * 8);
    ap.q[1] = *(const uint4*)(prow + 16 + half * 8);

    // O += P @ V  (V fragments already in flight since loop top)
#pragma unroll
    for (int ds = 0; ds < 4; ++ds)
      oacc[ds] = __builtin_amdgcn_wmma_f32_16x16x32_bf16(
          false, ap.bf, false, bv[ds].bf, (short)0, oacc[ds], false, false);
  }

  // finalize: divide by softmax denominator, store bf16 row-major
#pragma unroll
  for (int ds = 0; ds < 4; ++ds) {
#pragma unroll
    for (int r = 0; r < 8; ++r) {
      const float v = oacc[ds][r] / run_sum[r];
      const int gm = b * TDIM + q0 + r + half * 8;
      const int gn = h * HD + ds * 16 + ln;
      O[(size_t)gm * CDIM + gn] = f2bf_bits(v);
    }
  }
}

// ---------------------------------------------------------------- launch --
extern "C" void kernel_launch(void* const* d_in, const int* in_sizes, int n_in,
                              void* d_out, int out_size, void* d_ws, size_t ws_size,
                              hipStream_t stream) {
  const float* x       = (const float*)d_in[0];
  const float* w1_w    = (const float*)d_in[1];
  const float* w1_b    = (const float*)d_in[2];
  const float* w2      = (const float*)d_in[3];
  const float* b2      = (const float*)d_in[4];
  const float* value_w = (const float*)d_in[5];
  const float* value_b = (const float*)d_in[6];
  const float* proj_w  = (const float*)d_in[7];
  const float* proj_b  = (const float*)d_in[8];
  float* out = (float*)d_out;

  // workspace layout (bytes) — total 42 MB
  char* ws = (char*)d_ws;
  unsigned short* xb      = (unsigned short*)(ws);                       // 8 MB
  unsigned short* w1b     = (unsigned short*)(ws + ((size_t)8  << 20));  // 2 MB
  unsigned short* vwb     = (unsigned short*)(ws + ((size_t)10 << 20));  // 2 MB
  unsigned short* pwb     = (unsigned short*)(ws + ((size_t)12 << 20));  // 2 MB
  unsigned short* w2t     = (unsigned short*)(ws + ((size_t)14 << 20));  // 4 MB
  unsigned short* kbuf    = (unsigned short*)(ws + ((size_t)18 << 20));  // 8 MB
  unsigned short* vT      = (unsigned short*)(ws + ((size_t)26 << 20));  // 8 MB
  unsigned short* attnout = (unsigned short*)(ws + ((size_t)34 << 20));  // 8 MB

  const int nxc = BT * CDIM, ncc = CDIM * CDIM;
  cvt_f32_bf16<<<(nxc + 255) / 256, 256, 0, stream>>>(x, xb, nxc);
  cvt_f32_bf16<<<(ncc + 255) / 256, 256, 0, stream>>>(w1_w, w1b, ncc);
  cvt_f32_bf16<<<(ncc + 255) / 256, 256, 0, stream>>>(value_w, vwb, ncc);
  cvt_f32_bf16<<<(ncc + 255) / 256, 256, 0, stream>>>(proj_w, pwb, ncc);
  transpose_w2<<<(HDIM * TDIM * HD + 255) / 256, 256, 0, stream>>>(w2, w2t);

  dim3 ggrid(BT / 128, CDIM / 64);
  gemm_xWt<<<ggrid, 256, 0, stream>>>(xb, w1b, w1_b, kbuf, 0);   // k = relu
  gemm_xWt<<<ggrid, 256, 0, stream>>>(xb, vwb, value_b, vT, 1);  // v (transposed)

  synth_attn<<<dim3(TDIM / 64, HDIM, BDIM), 128, 0, stream>>>(kbuf, w2t, b2, vT,
                                                              attnout);

  gemm_xWt<<<ggrid, 256, 0, stream>>>(attnout, pwb, proj_b, out, 2);  // proj
}
